// DynamicRadiusChannelFusion_5471788335636
// MI455X (gfx1250) — matrix operand order repaired and checked by reading, not compiled, not run on hardware
//
#include <hip/hip_runtime.h>

#define B_    2
#define N_    8192
#define C_    128
#define KNN_K 16
#define TAUF  0.2f
#define BNEPS 1e-5f
#define MTOT  (B_ * N_)   // 16384
#define CHNK  2048        // candidate chunk for KNN (3*2048*4 = 24KB LDS)

typedef __attribute__((ext_vector_type(16))) __bf16 v16bf;
typedef __attribute__((ext_vector_type(8)))  float  v8f;
typedef __attribute__((ext_vector_type(4)))  int    v4i;

typedef __attribute__((address_space(1))) v4i* gptr_v4i;   // global
typedef __attribute__((address_space(3))) v4i* lptr_v4i;   // LDS

union Frag16 { v16bf v; uint4 q[2]; };

// ---- CDNA5 async global->LDS copy (ASYNCcnt path), guarded so absence still compiles
#if defined(__has_builtin)
#  if __has_builtin(__builtin_amdgcn_global_load_async_to_lds_b128)
#    define HAVE_ASYNC_LDS 1
#  endif
#endif
#ifndef HAVE_ASYNC_LDS
#  define HAVE_ASYNC_LDS 0
#endif

__device__ __forceinline__ void wait_async0() {
#if defined(__has_builtin)
#  if __has_builtin(__builtin_amdgcn_s_wait_asynccnt)
    __builtin_amdgcn_s_wait_asynccnt(0);
#  else
    asm volatile("s_wait_asynccnt 0x0" ::: "memory");
#  endif
#else
    asm volatile("s_wait_asynccnt 0x0" ::: "memory");
#endif
}

__device__ __forceinline__ void cp_async_b128(const unsigned short* gsrc, unsigned short* ldst) {
#if HAVE_ASYNC_LDS
    __builtin_amdgcn_global_load_async_to_lds_b128(
        (gptr_v4i)(unsigned short*)gsrc, (lptr_v4i)ldst, 0, 0);
#else
    *reinterpret_cast<uint4*>(ldst) = *reinterpret_cast<const uint4*>(gsrc);
#endif
}

__device__ __forceinline__ float n2n(float v) {
    unsigned u = __float_as_uint(v);
    if (((u >> 23) & 0xFFu) == 0xFFu)
        return (u & 0x7FFFFFu) ? 0.0f : ((u >> 31) ? -1.0e4f : 1.0e4f);
    return v;
}

__device__ __forceinline__ unsigned short f2bf(float f) {
    unsigned u = __float_as_uint(f);
    u += 0x7FFFu + ((u >> 16) & 1u);   // round-to-nearest-even
    return (unsigned short)(u >> 16);
}

// ---------------------------------------------------------------- init stats
__global__ void k_init(float* stats) {
    int t = blockIdx.x * blockDim.x + threadIdx.x;
    if (t < 1024) stats[t] = 0.0f;
}

// ---------------------------------------------------------------- f32 -> bf16 weight conversion
__global__ void k_cvt(const float* __restrict__ src, unsigned short* __restrict__ dst, int n) {
    int t = blockIdx.x * 256 + threadIdx.x;
    if (t < n) dst[t] = f2bf(src[t]);
}

// ------------------------------------------------- x [B,C,N] -> xt [B,N,C] + nan_to_num
__global__ __launch_bounds__(256) void k_transpose_x(const float* __restrict__ x,
                                                     float* __restrict__ xt) {
    __shared__ float tile[32][33];
    int b = blockIdx.z;
    int c0 = blockIdx.y * 32, n0 = blockIdx.x * 32;
    int tx = threadIdx.x, ty = threadIdx.y;
#pragma unroll
    for (int i = 0; i < 32; i += 8)
        tile[ty + i][tx] = n2n(x[((size_t)(b * C_ + c0 + ty + i)) * N_ + n0 + tx]);
    __syncthreads();
#pragma unroll
    for (int i = 0; i < 32; i += 8)
        xt[((size_t)(b * N_ + n0 + ty + i)) * C_ + c0 + tx] = tile[tx][ty + i];
}

// ---------------------------------------------------------------- brute-force KNN + softmax weights
__global__ __launch_bounds__(256) void k_knn(const float* __restrict__ pos,
                                             int* __restrict__ knn_idx,
                                             float* __restrict__ knn_w) {
    __shared__ float sx[CHNK], sy[CHNK], sz[CHNK];
    int b = blockIdx.y;
    int q = blockIdx.x * 256 + threadIdx.x;
    const float* pb = pos + (size_t)b * 3 * N_;
    float qx = n2n(pb[q]), qy = n2n(pb[N_ + q]), qz = n2n(pb[2 * N_ + q]);

    float bd[KNN_K]; int bi[KNN_K];
#pragma unroll
    for (int t = 0; t < KNN_K; ++t) { bd[t] = 3.4e38f; bi[t] = 0; }

    for (int c0 = 0; c0 < N_; c0 += CHNK) {
        __syncthreads();
        for (int t = threadIdx.x; t < CHNK; t += 256) {
            sx[t] = n2n(pb[c0 + t]);
            sy[t] = n2n(pb[N_ + c0 + t]);
            sz[t] = n2n(pb[2 * N_ + c0 + t]);
        }
        __syncthreads();
        for (int j = 0; j < CHNK; ++j) {
            float dx = qx - sx[j], dy = qy - sy[j], dz = qz - sz[j];
            float d2 = dx * dx + dy * dy + dz * dz;
            if (d2 < bd[KNN_K - 1]) {          // sorted-register insertion
                float cd = d2; int ci = c0 + j;
#pragma unroll
                for (int t = 0; t < KNN_K; ++t) {
                    bool sw = cd < bd[t];
                    float td = sw ? bd[t] : cd; int ti = sw ? bi[t] : ci;
                    bd[t] = sw ? cd : bd[t];    bi[t] = sw ? ci : bi[t];
                    cd = td; ci = ti;
                }
            }
        }
    }
    float dmin = fmaxf(sqrtf(bd[0]), 1e-6f);
    float ev[KNN_K]; float ssum = 0.0f;
#pragma unroll
    for (int t = 0; t < KNN_K; ++t) {
        float dt = fmaxf(sqrtf(bd[t]), 1e-6f);
        ev[t] = expf((dmin - dt) * (1.0f / TAUF));
        ssum += ev[t];
    }
    float rs = 1.0f / ssum;
#pragma unroll
    for (int t = 0; t < KNN_K; ++t) {
        knn_idx[(size_t)(b * N_ + q) * KNN_K + t] = bi[t];
        knn_w [(size_t)(b * N_ + q) * KNN_K + t] = ev[t] * rs;
    }
}

// ---------------------------------------------------------------- weighted aggregate + concat (writes bf16)
__global__ __launch_bounds__(128) void k_fuse(const float* __restrict__ xt,
                                              const int* __restrict__ knn_idx,
                                              const float* __restrict__ knn_w,
                                              unsigned short* __restrict__ fuse) {
    int c = threadIdx.x;
    int q0 = blockIdx.x * 8;
    for (int j = 0; j < 8; ++j) {
        int q = q0 + j;
        int b = q >> 13;               // N_ = 8192
        float acc = 0.0f;
#pragma unroll
        for (int k = 0; k < KNN_K; ++k) {
            int   nb = knn_idx[(size_t)q * KNN_K + k];
            float wk = knn_w [(size_t)q * KNN_K + k];
            acc = fmaf(wk, xt[((size_t)(b * N_) + nb) * C_ + c], acc);
        }
        fuse[(size_t)q * (2 * C_) + c]      = f2bf(xt[(size_t)q * C_ + c]);
        fuse[(size_t)q * (2 * C_) + C_ + c] = f2bf(acc);
    }
}

// ---------------------------------------------------------------- WMMA bf16 GEMM: Out[M,128] = A[M,KTOT] * W[128,KTOT]^T
// A, W already bf16 in global memory -> staging is a pure async copy (ASYNCcnt path).
template <int KTOT>
__global__ __launch_bounds__(256) void k_gemm(const unsigned short* __restrict__ A,
                                              const unsigned short* __restrict__ W,
                                              float* __restrict__ Out) {
    __shared__ unsigned short As[128 * 72];   // 64-wide K chunk, 72-ushort row stride (16B aligned, bank-spread)
    __shared__ unsigned short Ws[128 * 72];
    const int tid = threadIdx.x;
    const int l  = tid & 31;
    const int wv = tid >> 5;
    const int lm = l & 15, lh = l >> 4;
    const int mblock = blockIdx.x * 128;
    const int m0 = wv * 16;
    v8f acc[8] = {};

    for (int kc = 0; kc < KTOT; kc += 64) {
        __syncthreads();
#pragma unroll
        for (int g = 0; g < 4; ++g) {          // 1024 16-byte chunks: 128 rows x 8 chunks each for A and W
            int lin = g * 256 + tid;
            int r = lin >> 3, cc = lin & 7;    // row, 16B chunk within 64-ushort row
            cp_async_b128(A + (size_t)(mblock + r) * KTOT + kc + cc * 8, &As[r * 72 + cc * 8]);
            cp_async_b128(W + (size_t)r * KTOT + kc + cc * 8,            &Ws[r * 72 + cc * 8]);
        }
#if HAVE_ASYNC_LDS
        wait_async0();
#endif
        __syncthreads();
        if (kc + 64 < KTOT)
            __builtin_prefetch(A + (size_t)(mblock + (tid >> 1)) * KTOT + kc + 64, 0, 1);

        const unsigned short* ar = &As[(m0 + lm) * 72];
#pragma unroll
        for (int ks = 0; ks < 64; ks += 32) {
            Frag16 af;   // A 16x32 bf16: lane m=l&15; K = 8*(l>>4)+{0..7}, +16..23
            af.q[0] = *reinterpret_cast<const uint4*>(ar + ks + lh * 8);
            af.q[1] = *reinterpret_cast<const uint4*>(ar + ks + lh * 8 + 16);
            Frag16 bcur;  // B 32x16 bf16: lane n=l&15; K = 8*(l>>4)+{0..7}, +16..23
            {
                const unsigned short* wr = &Ws[lm * 72] + ks + lh * 8;
                bcur.q[0] = *reinterpret_cast<const uint4*>(wr);
                bcur.q[1] = *reinterpret_cast<const uint4*>(wr + 16);
            }
#pragma unroll
            for (int t = 0; t < 8; ++t) {
                Frag16 bnext;   // software pipeline: load t+1 before WMMA t
                if (t < 7) {
                    const unsigned short* wr = &Ws[((t + 1) * 16 + lm) * 72] + ks + lh * 8;
                    bnext.q[0] = *reinterpret_cast<const uint4*>(wr);
                    bnext.q[1] = *reinterpret_cast<const uint4*>(wr + 16);
                }
                acc[t] = __builtin_amdgcn_wmma_f32_16x16x32_bf16(
                    false, af.v, false, bcur.v, (short)0, acc[t], false, false);
                if (t < 7) bcur = bnext;
            }
        }
    }
#pragma unroll
    for (int t = 0; t < 8; ++t) {
        int n = t * 16 + lm;
#pragma unroll
        for (int i = 0; i < 8; ++i) {          // C/D: lane l, vgpr i -> M = 8*(l>>4)+i, N = l&15
            int m = mblock + m0 + lh * 8 + i;
            Out[(size_t)m * 128 + n] = acc[t][i];
        }
    }
}

// ---------------------------------------------------------------- per-channel sum / sumsq
__global__ __launch_bounds__(128) void k_stats(const float* __restrict__ pre,
                                               float* __restrict__ sum, float* __restrict__ sq) {
    int c = threadIdx.x;
    int r0 = blockIdx.x * 256;
    float s = 0.0f, s2 = 0.0f;
    for (int i = 0; i < 256; ++i) {
        float v = pre[(size_t)(r0 + i) * 128 + c];
        s += v; s2 += v * v;
    }
    atomicAdd(&sum[c], s);
    atomicAdd(&sq[c], s2);
}

// ---------------------------------------------------------------- BN + ReLU (bf16 and/or f32 out, optional SE pooling)
__global__ __launch_bounds__(128) void k_bnrelu(const float* __restrict__ pre,
                                                const float* __restrict__ sum, const float* __restrict__ sq,
                                                const float* __restrict__ gamma, const float* __restrict__ beta,
                                                float* __restrict__ outf, unsigned short* __restrict__ outb,
                                                float* __restrict__ pool) {
    int c = threadIdx.x;
    int r0 = blockIdx.x * 16;
    const float inv = 1.0f / (float)MTOT;
    float mean = sum[c] * inv;
    float var  = fmaxf(sq[c] * inv - mean * mean, 0.0f);
    float sc = rsqrtf(var + BNEPS) * gamma[c];
    float sh = beta[c] - mean * sc;
    float acc = 0.0f;
    for (int i = 0; i < 16; ++i) {
        float v = fmaxf(pre[(size_t)(r0 + i) * 128 + c] * sc + sh, 0.0f);
        if (outf) outf[(size_t)(r0 + i) * 128 + c] = v;
        if (outb) outb[(size_t)(r0 + i) * 128 + c] = f2bf(v);
        acc += v;
    }
    if (pool) atomicAdd(&pool[(r0 >> 13) * 128 + c], acc);
}

// ---------------------------------------------------------------- SE gate
__global__ __launch_bounds__(256) void k_se(const float* __restrict__ pool,
                                            const float* __restrict__ fc1w, const float* __restrict__ fc1b,
                                            const float* __restrict__ fc2w, const float* __restrict__ fc2b,
                                            float* __restrict__ gate) {
    __shared__ float s[2][128];
    __shared__ float hid[2][8];
    int tid = threadIdx.x;
    { int b = tid >> 7, c = tid & 127; s[b][c] = pool[tid] * (1.0f / (float)N_); }
    __syncthreads();
    if (tid < 16) {
        int b = tid >> 3, j = tid & 7;
        float a = fc1b[j];
        for (int c = 0; c < 128; ++c) a = fmaf(s[b][c], fc1w[j * 128 + c], a);
        hid[b][j] = fmaxf(a, 0.0f);
    }
    __syncthreads();
    { int b = tid >> 7, c = tid & 127;
      float a = fc2b[c];
#pragma unroll
      for (int j = 0; j < 8; ++j) a = fmaf(hid[b][j], fc2w[c * 8 + j], a);
      gate[tid] = 1.0f / (1.0f + expf(-a)); }
}

// ---------------------------------------------------------------- residual + gate + transpose back to [B,C,N]
__global__ __launch_bounds__(256) void k_final(const float* __restrict__ xt,
                                               const float* __restrict__ y,
                                               const float* __restrict__ gate,
                                               float* __restrict__ out) {
    __shared__ float tile[32][33];
    int b = blockIdx.z;
    int c0 = blockIdx.y * 32, n0 = blockIdx.x * 32;
    int tx = threadIdx.x, ty = threadIdx.y;
    float g = gate[b * C_ + c0 + tx];
#pragma unroll
    for (int i = 0; i < 32; i += 8) {
        size_t row = (size_t)(b * N_ + n0 + ty + i) * C_;
        tile[tx][ty + i] = xt[row + c0 + tx] + y[row + c0 + tx] * g;
    }
    __syncthreads();
#pragma unroll
    for (int i = 0; i < 32; i += 8)
        out[((size_t)(b * C_ + c0 + ty + i)) * N_ + n0 + tx] = tile[ty + i][tx];
}

// ================================================================ host launch
extern "C" void kernel_launch(void* const* d_in, const int* in_sizes, int n_in,
                              void* d_out, int out_size, void* d_ws, size_t ws_size,
                              hipStream_t stream) {
    const float* x    = (const float*)d_in[0];
    const float* pos  = (const float*)d_in[1];
    const float* w1   = (const float*)d_in[2];
    const float* g1   = (const float*)d_in[3];
    const float* b1   = (const float*)d_in[4];
    const float* w2   = (const float*)d_in[5];
    const float* g2   = (const float*)d_in[6];
    const float* b2   = (const float*)d_in[7];
    const float* fc1w = (const float*)d_in[8];
    const float* fc1b = (const float*)d_in[9];
    const float* fc2w = (const float*)d_in[10];
    const float* fc2b = (const float*)d_in[11];
    float* out = (float*)d_out;

    // workspace layout (float units); y (f32) aliases the fuse_bf16 region (dead after gemm1)
    float* ws = (float*)d_ws;
    float*          xt    = ws;                          // 2,097,152 f
    int*            kidx  = (int*)(ws + 2097152);        //   262,144
    float*          kw    = ws + 2359296;                //   262,144
    unsigned short* fuse  = (unsigned short*)(ws + 2621440); // 4,194,304 ushort (2,097,152 f-slots)
    float*          yv    = ws + 2621440;                // alias of fuse region, 2,097,152 f
    unsigned short* hv    = (unsigned short*)(ws + 4718592); // 2,097,152 ushort (1,048,576 f-slots)
    float*          pre   = ws + 5767168;                // 2,097,152 f
    unsigned short* w1b   = (unsigned short*)(ws + 7864320); // 32,768 ushort
    unsigned short* w2b   = (unsigned short*)(ws + 7880704); // 16,384 ushort
    float*          stats = ws + 7888896;                // 1,024 f
    float* sum1 = stats;        float* sq1  = stats + 128;
    float* sum2 = stats + 256;  float* sq2  = stats + 384;
    float* pool = stats + 512;  float* gate = stats + 768;

    k_init<<<4, 256, 0, stream>>>(stats);
    k_cvt<<<128, 256, 0, stream>>>(w1, w1b, 128 * 256);
    k_cvt<<<64, 256, 0, stream>>>(w2, w2b, 128 * 128);
    k_transpose_x<<<dim3(N_ / 32, C_ / 32, B_), dim3(32, 8), 0, stream>>>(x, xt);
    k_knn<<<dim3(N_ / 256, B_), 256, 0, stream>>>(pos, kidx, kw);
    k_fuse<<<MTOT / 8, 128, 0, stream>>>(xt, kidx, kw, fuse);
    k_gemm<256><<<MTOT / 128, 256, 0, stream>>>(fuse, w1b, pre);
    k_stats<<<MTOT / 256, 128, 0, stream>>>(pre, sum1, sq1);
    k_bnrelu<<<MTOT / 16, 128, 0, stream>>>(pre, sum1, sq1, g1, b1, nullptr, hv, nullptr);
    k_gemm<128><<<MTOT / 128, 256, 0, stream>>>(hv, w2b, pre);
    k_stats<<<MTOT / 256, 128, 0, stream>>>(pre, sum2, sq2);
    k_bnrelu<<<MTOT / 16, 128, 0, stream>>>(pre, sum2, sq2, g2, b2, yv, nullptr, pool);
    k_se<<<1, 256, 0, stream>>>(pool, fc1w, fc1b, fc2w, fc2b, gate);
    k_final<<<dim3(N_ / 32, C_ / 32, B_), dim3(32, 8), 0, stream>>>(xt, yv, gate, out);
}
